// LSTM_66743791780055
// MI455X (gfx1250) — compile-verified, matrix operand order checked by reading, not run on hardware
//
#include <hip/hip_runtime.h>

// ---------------------------------------------------------------------------
// LSTM on MI455X (gfx1250, wave32, WMMA)
//   T=512, B=64, D=128, H=256, gates = 4H = 1024
//
// Strategy:
//   k0: pack Wall=[Wf|Wi|Wc|Wo] ([384 x 1024]) into per-lane WMMA B-fragment
//       order, split fp32 -> bf16 hi + bf16 lo (bf16x3 precision trick).
//   k1: Xproj[t,b,:] = x @ Wx + bias  (parallel GEMM, wmma bf16, fp32 acc)
//   k2: 8 persistent WGs, Wh slice resident in LDS (128KB bf16 hi/lo),
//       per-step [64x256]@[256x128] wmma GEMM + gate math; global spin
//       barrier + double-buffered packed h fragments between steps.
// ---------------------------------------------------------------------------

typedef __attribute__((ext_vector_type(16))) __bf16 v16bf;
typedef __attribute__((ext_vector_type(8)))  float  v8f;

#define T_   512
#define B_   64
#define D_   128
#define H_   256
#define NG_  1024   // 4*H
#define NWG_ 8      // recurrent workgroups (each owns 32 h-columns)

// workspace layout (bytes)
#define OFF_SYNC      0          // 256 B   (barrier counter)
#define OFF_APACK_HI  256        // 65536 B ([2 buf][4 m][8 kt][32 lane][16] bf16)
#define OFF_APACK_LO  65792      // 65536 B
#define OFF_WPACK_HI  131328     // 786432 B ([64 NT][12 kt][32 lane][16] bf16)
#define OFF_WPACK_LO  917760     // 786432 B
#define OFF_XPROJ     1704192    // 512*64*1024*4 = 134217728 B
#define WS_NEEDED     (1704192ull + 134217728ull)

__device__ __forceinline__ float sigmoidf_(float v) {
    return 1.0f / (1.0f + __expf(-v));
}

// fragment K index for 16-bit WMMA operands (ISA 7.12.2):
//   k(e, lane) = (e<8 ? e : e+8) + (lane>>4)*8

// ---------------------------------------------------------------------------
// k0: pack weights into WMMA B-fragment order, bf16 hi/lo split.
//   Wpack[NT][kt][lane][e]; NT = global N-tile (0..63), kt = K-tile of 32
//   (kt 0..7 -> rows 0..255 = h part; kt 8..11 -> rows 256..383 = x part)
// ---------------------------------------------------------------------------
__global__ void __launch_bounds__(256)
pack_weights(const float* __restrict__ Wf, const float* __restrict__ Wi,
             const float* __restrict__ Wc, const float* __restrict__ Wo,
             __bf16* __restrict__ wh, __bf16* __restrict__ wl) {
    int idx = blockIdx.x * 256 + threadIdx.x;        // [0, 64*12*32*16)
    if (idx >= 64 * 12 * 32 * 16) return;
    int e    = idx & 15;
    int lane = (idx >> 4) & 31;
    int kt   = (idx >> 9) % 12;
    int NT   = (idx >> 9) / 12;
    int n    = NT * 16 + (lane & 15);                // gate column 0..1023
    int gate = n >> 8;
    int j    = n & 255;
    int klocal = ((e < 8) ? e : e + 8) + ((lane >> 4) * 8);
    int row  = kt * 32 + klocal;                     // 0..383 row of Wall
    const float* W = (gate == 0) ? Wf : (gate == 1) ? Wi : (gate == 2) ? Wc : Wo;
    float v = W[row * H_ + j];
    __bf16 hi = (__bf16)v;
    wh[idx] = hi;
    wl[idx] = (__bf16)(v - (float)hi);
}

// ---------------------------------------------------------------------------
// k1: Xproj = x @ Wx + bias.  M = T*B = 32768, N = 1024, K = 128.
//   block = 256 thr (8 waves); block tile = 64(M) x 128(N);
//   wave (mt = w>>1, half = w&1) owns 16x64 = 4 C tiles.
// ---------------------------------------------------------------------------
__global__ void __launch_bounds__(256)
xproj_kernel(const float* __restrict__ x,
             const __bf16* __restrict__ wh, const __bf16* __restrict__ wl,
             const float* __restrict__ bf, const float* __restrict__ bi,
             const float* __restrict__ bc, const float* __restrict__ bo,
             float* __restrict__ xproj) {
    const int lane = threadIdx.x & 31;
    const int wv   = threadIdx.x >> 5;
    const int mt   = wv >> 1;
    const int half = wv & 1;
    const int mblk = blockIdx.x;      // 0..511  (M tiles of 64)
    const int nblk = blockIdx.y;      // 0..7    (N tiles of 128)

    const int Mrow  = mblk * 64 + mt * 16 + (lane & 15);
    const int kbase = (lane >> 4) * 8;

    v8f acc[4] = {{0,0,0,0,0,0,0,0},{0,0,0,0,0,0,0,0},
                  {0,0,0,0,0,0,0,0},{0,0,0,0,0,0,0,0}};

    for (int ktx = 0; ktx < 4; ++ktx) {
        // A fragment from fp32 x, split to bf16 hi/lo
        const float* xp = x + (size_t)Mrow * D_ + ktx * 32 + kbase;
        float va[16] __attribute__((aligned(16)));
        *(float4*)&va[0]  = *(const float4*)(xp + 0);
        *(float4*)&va[4]  = *(const float4*)(xp + 4);
        *(float4*)&va[8]  = *(const float4*)(xp + 16);
        *(float4*)&va[12] = *(const float4*)(xp + 20);
        v16bf ah, al;
        #pragma unroll
        for (int e = 0; e < 16; ++e) {
            __bf16 hv = (__bf16)va[e];
            ah[e] = hv;
            al[e] = (__bf16)(va[e] - (float)hv);
        }
        const int kt = 8 + ktx;   // x-part rows of Wall
        #pragma unroll
        for (int q = 0; q < 4; ++q) {
            const int NT = nblk * 8 + half * 4 + q;
            const size_t boff = (((size_t)NT * 12 + kt) * 32 + lane) * 16;
            v16bf bh = *(const v16bf*)(wh + boff);
            v16bf bl = *(const v16bf*)(wl + boff);
            acc[q] = __builtin_amdgcn_wmma_f32_16x16x32_bf16(false, ah, false, bh, (short)0, acc[q], false, false);
            acc[q] = __builtin_amdgcn_wmma_f32_16x16x32_bf16(false, ah, false, bl, (short)0, acc[q], false, false);
            acc[q] = __builtin_amdgcn_wmma_f32_16x16x32_bf16(false, al, false, bh, (short)0, acc[q], false, false);
        }
    }
    // epilogue: + bias, store fp32
    #pragma unroll
    for (int q = 0; q < 4; ++q) {
        const int NT   = nblk * 8 + half * 4 + q;
        const int N    = NT * 16 + (lane & 15);
        const int gate = N >> 8;
        const int j    = N & 255;
        const float* bp = (gate == 0) ? bf : (gate == 1) ? bi : (gate == 2) ? bc : bo;
        const float bias = bp[j];
        const int Mb = mblk * 64 + mt * 16 + (lane >> 4) * 8;
        #pragma unroll
        for (int v = 0; v < 8; ++v)
            xproj[(size_t)(Mb + v) * NG_ + N] = acc[q][v] + bias;
    }
}

// ---------------------------------------------------------------------------
// k2: persistent recurrent kernel. 8 WGs x 256 thr. WG w owns h columns
//   [w*32, w*32+32) i.e. gate columns {g*256 + w*32 + j : g<4, j<32}.
// ---------------------------------------------------------------------------
__global__ void __launch_bounds__(256)
lstm_kernel(const float* __restrict__ h0, const float* __restrict__ c0,
            const __bf16* __restrict__ wh, const __bf16* __restrict__ wl,
            const float* __restrict__ xproj,
            __bf16* __restrict__ aph, __bf16* __restrict__ apl,
            unsigned* __restrict__ syncc, float* __restrict__ out) {
    __shared__ __align__(32) __bf16 sB[2][8][8][32][16];  // 128 KB weights hi/lo
    __shared__ float sG[4][32][64];                       // 32 KB gate pre-acts
    __shared__ float sC[32][64];                          // 8 KB c state [j][b]
    __shared__ float sH[32][64];                          // 8 KB h_new   [j][b]

    const int tid  = threadIdx.x;
    const int lane = tid & 31;
    const int wv   = tid >> 5;
    const int mt   = wv >> 1;      // 0..3  M tile (16 batch rows)
    const int half = wv & 1;       // 0..1  which 4 of the 8 local N tiles
    const int w    = blockIdx.x;   // 0..7  column-slice owner

    // --- stage Wh slice into LDS (fragment-ordered, plain vector copies) ---
    for (int job = tid; job < 4096; job += 256) {   // [2][8][8][32] rows of 32B
        int l   = job & 31;
        int kt  = (job >> 5) & 7;
        int ntl = (job >> 8) & 7;
        int hl  = job >> 11;
        int NT  = (ntl >> 1) * 16 + w * 2 + (ntl & 1);
        const uint4* src = (const uint4*)((hl ? wl : wh) +
                           (((size_t)NT * 12 + kt) * 32 + l) * 16);
        uint4* dst = (uint4*)&sB[hl][ntl][kt][l][0];
        dst[0] = src[0];
        dst[1] = src[1];
    }
    // --- init c slice in LDS ---
    for (int pr = tid; pr < 2048; pr += 256) {
        int j = pr >> 6, b = pr & 63;
        sC[j][b] = c0[b * H_ + w * 32 + j];
    }
    // --- init packed A fragments (buffer 0) from h0; WG w fills ktile w ---
    if (tid < 128) {
        int m = tid >> 5, l = tid & 31;
        int b  = m * 16 + (l & 15);
        int j0 = (l >> 4) * 8;
        v16bf hh, ll;
        #pragma unroll
        for (int e = 0; e < 8; ++e) {
            float v0 = h0[b * H_ + w * 32 + j0 + e];
            float v1 = h0[b * H_ + w * 32 + j0 + 16 + e];
            __bf16 t0 = (__bf16)v0, t1 = (__bf16)v1;
            hh[e] = t0; hh[e + 8] = t1;
            ll[e] = (__bf16)(v0 - (float)t0);
            ll[e + 8] = (__bf16)(v1 - (float)t1);
        }
        size_t off = (((size_t)m * 8 + w) * 32 + l) * 16;   // buf 0
        *(v16bf*)(aph + off) = hh;
        *(v16bf*)(apl + off) = ll;
    }

    const int Mb = mt * 16 + (lane >> 4) * 8;

    for (int t = 0; t < T_; ++t) {
        // ---- device-wide barrier (release my stores, acquire others') ----
        __threadfence();
        __syncthreads();
        if (tid == 0) {
            atomicAdd(syncc, 1u);
            const unsigned target = (unsigned)(t + 1) * NWG_;
            while (*(volatile unsigned*)syncc < target)
                __builtin_amdgcn_s_sleep(1);
            __threadfence();
        }
        __syncthreads();

        const int p = t & 1;     // A-fragment buffer to read

        // ---- acc init from precomputed Xproj (includes bias) ----
        v8f acc[4];
        #pragma unroll
        for (int q = 0; q < 4; ++q) {
            const int ntl = half * 4 + q;
            const int N = (ntl >> 1) * 256 + w * 32 + (ntl & 1) * 16 + (lane & 15);
            #pragma unroll
            for (int v = 0; v < 8; ++v)
                acc[q][v] = xproj[((size_t)t * 64 + Mb + v) * NG_ + N];
        }

        // ---- gates += h @ Wh  (bf16x3: hi*hi + hi*lo + lo*hi) ----
        for (int kt = 0; kt < 8; ++kt) {
            const size_t aoff = (((size_t)(p * 4 + mt) * 8 + kt) * 32 + lane) * 16;
            v16bf ah = *(const v16bf*)(aph + aoff);
            v16bf al = *(const v16bf*)(apl + aoff);
            #pragma unroll
            for (int q = 0; q < 4; ++q) {
                const int ntl = half * 4 + q;
                v16bf bh = *(const v16bf*)&sB[0][ntl][kt][lane][0];
                v16bf bl = *(const v16bf*)&sB[1][ntl][kt][lane][0];
                acc[q] = __builtin_amdgcn_wmma_f32_16x16x32_bf16(false, ah, false, bh, (short)0, acc[q], false, false);
                acc[q] = __builtin_amdgcn_wmma_f32_16x16x32_bf16(false, ah, false, bl, (short)0, acc[q], false, false);
                acc[q] = __builtin_amdgcn_wmma_f32_16x16x32_bf16(false, al, false, bh, (short)0, acc[q], false, false);
            }
        }

        // ---- scatter gate pre-activations to LDS ----
        #pragma unroll
        for (int q = 0; q < 4; ++q) {
            const int ntl  = half * 4 + q;
            const int gate = ntl >> 1;
            const int j    = (ntl & 1) * 16 + (lane & 15);
            #pragma unroll
            for (int v = 0; v < 8; ++v)
                sG[gate][j][Mb + v] = acc[q][v];
        }
        __syncthreads();

        // ---- elementwise gate math; write outputs; update c in LDS ----
        for (int pr = tid; pr < 2048; pr += 256) {
            int j = pr >> 6, b = pr & 63;
            float fg = sigmoidf_(sG[0][j][b]);
            float ig = sigmoidf_(sG[1][j][b]);
            float gg = tanhf(sG[2][j][b]);
            float og = sigmoidf_(sG[3][j][b]);
            float cn = fg * sC[j][b] + ig * gg;
            float hn = og * tanhf(cn);
            sC[j][b] = cn;
            sH[j][b] = hn;
            size_t o = ((size_t)t * B_ + b) * H_ + w * 32 + j;
            out[o] = hn;                                    // hiddens
            out[(size_t)T_ * B_ * H_ + o] = cn;             // contexts
        }
        __syncthreads();

        // ---- pack h_new -> A-fragment buffer p^1 (this WG's ktile = w) ----
        if (tid < 128) {
            int m = tid >> 5, l = tid & 31;
            int b  = m * 16 + (l & 15);
            int j0 = (l >> 4) * 8;
            v16bf hh, ll;
            #pragma unroll
            for (int e = 0; e < 8; ++e) {
                float v0 = sH[j0 + e][b];
                float v1 = sH[j0 + 16 + e][b];
                __bf16 t0 = (__bf16)v0, t1 = (__bf16)v1;
                hh[e] = t0; hh[e + 8] = t1;
                ll[e] = (__bf16)(v0 - (float)t0);
                ll[e + 8] = (__bf16)(v1 - (float)t1);
            }
            size_t off = (((size_t)((p ^ 1) * 4 + m) * 8 + w) * 32 + l) * 16;
            *(v16bf*)(aph + off) = hh;
            *(v16bf*)(apl + off) = ll;
        }
    }
}

// ---------------------------------------------------------------------------
extern "C" void kernel_launch(void* const* d_in, const int* in_sizes, int n_in,
                              void* d_out, int out_size, void* d_ws, size_t ws_size,
                              hipStream_t stream) {
    (void)in_sizes; (void)n_in; (void)out_size;
    if (ws_size < WS_NEEDED) return;   // workspace too small: bail safely

    const float* x  = (const float*)d_in[0];
    const float* h0 = (const float*)d_in[1];
    const float* c0 = (const float*)d_in[2];
    const float* Wf = (const float*)d_in[3];
    const float* bf = (const float*)d_in[4];
    const float* Wi = (const float*)d_in[5];
    const float* bi = (const float*)d_in[6];
    const float* Wc = (const float*)d_in[7];
    const float* bc = (const float*)d_in[8];
    const float* Wo = (const float*)d_in[9];
    const float* bo = (const float*)d_in[10];

    char*     ws    = (char*)d_ws;
    unsigned* syncc = (unsigned*)(ws + OFF_SYNC);
    __bf16*   aph   = (__bf16*)(ws + OFF_APACK_HI);
    __bf16*   apl   = (__bf16*)(ws + OFF_APACK_LO);
    __bf16*   wph   = (__bf16*)(ws + OFF_WPACK_HI);
    __bf16*   wpl   = (__bf16*)(ws + OFF_WPACK_LO);
    float*    xpj   = (float*)(ws + OFF_XPROJ);
    float*    out   = (float*)d_out;

    hipMemsetAsync(syncc, 0, 256, stream);                       // barrier ctr
    pack_weights<<<1536, 256, 0, stream>>>(Wf, Wi, Wc, Wo, wph, wpl);
    xproj_kernel<<<dim3(512, 8), 256, 0, stream>>>(x, wph, wpl, bf, bi, bc, bo, xpj);
    lstm_kernel<<<NWG_, 256, 0, stream>>>(h0, c0, wph, wpl, xpj, aph, apl, syncc, out);
}
